// NODERegressor_2860448219731
// MI455X (gfx1250) — matrix operand order repaired and checked by reading, not compiled, not run on hardware
//
#include <hip/hip_runtime.h>

#define BATCH       2048
#define IN_F        64
#define NUM_TREES   128
#define DEPTH       6
#define NUM_LEAVES  64
#define NUM_LAYERS  2
#define NCOLS       (NUM_LAYERS * NUM_TREES * DEPTH)   // 1536
#define TREES_TOTAL (NUM_LAYERS * NUM_TREES)           // 256
#define BN_EPS      1e-5f

typedef __attribute__((ext_vector_type(16))) _Float16 v16h;
typedef __attribute__((ext_vector_type(8)))  _Float16 v8h;
typedef __attribute__((ext_vector_type(8)))  float    v8f;

// ---------------------------------------------------------------------------
// Kernel 1: BatchNorm (training-mode batch stats) -> f16 x_hat (A matrix)
// One block per feature column. 256 threads stride over the 2048 rows.
// ---------------------------------------------------------------------------
__global__ void bn_kernel(const float* __restrict__ x,
                          const float* __restrict__ gamma,
                          const float* __restrict__ beta,
                          _Float16* __restrict__ xhat) {
    const int f   = blockIdx.x;       // 0..63
    const int tid = threadIdx.x;      // 0..255

    float s = 0.0f, sq = 0.0f;
    #pragma unroll
    for (int r = tid; r < BATCH; r += 256) {
        float v = x[(size_t)r * IN_F + f];
        s  += v;
        sq += v * v;
    }

    __shared__ float ss[256];
    __shared__ float sq2[256];
    ss[tid] = s; sq2[tid] = sq;
    __syncthreads();
    for (int st = 128; st > 0; st >>= 1) {
        if (tid < st) { ss[tid] += ss[tid + st]; sq2[tid] += sq2[tid + st]; }
        __syncthreads();
    }

    __shared__ float mean_s, scale_s, shift_s;
    if (tid == 0) {
        float mean = ss[0]  * (1.0f / (float)BATCH);
        float var  = sq2[0] * (1.0f / (float)BATCH) - mean * mean;  // biased
        float rstd = rsqrtf(var + BN_EPS);
        mean_s  = mean;
        scale_s = rstd * gamma[f];
        shift_s = beta[f];
    }
    __syncthreads();

    const float mean = mean_s, sc = scale_s, sh = shift_s;
    for (int r = tid; r < BATCH; r += 256) {
        float v = (x[(size_t)r * IN_F + f] - mean) * sc + sh;
        xhat[(size_t)r * IN_F + f] = (_Float16)v;
    }
}

// ---------------------------------------------------------------------------
// Kernel 2: softmax over IN_F for each (layer,tree,depth) row -> f16 gates.
// One wave32 per row; each lane owns 2 of the 64 elements.
// ---------------------------------------------------------------------------
__global__ void softmax_kernel(const float* __restrict__ fs,
                               _Float16* __restrict__ gatesH) {
    const int lane = threadIdx.x & 31;
    const int row  = blockIdx.x * 8 + (threadIdx.x >> 5);   // 1536 rows total

    const float* rp = fs + (size_t)row * IN_F;
    float v0 = rp[lane];
    float v1 = rp[lane + 32];

    float m = fmaxf(v0, v1);
    #pragma unroll
    for (int off = 16; off > 0; off >>= 1)
        m = fmaxf(m, __shfl_xor(m, off, 32));

    float e0 = __expf(v0 - m);
    float e1 = __expf(v1 - m);
    float s = e0 + e1;
    #pragma unroll
    for (int off = 16; off > 0; off >>= 1)
        s += __shfl_xor(s, off, 32);

    float inv = 1.0f / s;
    _Float16* op = gatesH + (size_t)row * IN_F;
    op[lane]      = (_Float16)(e0 * inv);
    op[lane + 32] = (_Float16)(e1 * inv);
}

// ---------------------------------------------------------------------------
// Kernel 3: x_proj = x_hat(2048x64) * gates^T(64x1536) via v_wmma_f32_16x16x32_f16.
// One wave per 16x16 output tile; K=64 -> two WMMA ops. Grid exactly covers all
// tiles so EXEC is all-ones (WMMA requirement).
//
// Fragment layouts (CDNA5 ISA 7.12.2, wave32):
//   A (16x32 f16): lane m=L&15; lanes 0-15 hold K 0..7 & 16..23, lanes 16-31
//                  hold K 8..15 & 24..31 (two 8-half contiguous segments).
//   B (32x16 f16): lane col n=L&15; lanes 0-15 hold K 0..7 & 16..23, lanes 16-31
//                  hold K 8..15 & 24..31 (same split as A).
//   C/D (16x16 f32): lane col n=L&15; VGPR r -> row (L>>4)*8 + r.
// ---------------------------------------------------------------------------
__global__ void gemm_kernel(const _Float16* __restrict__ A,   // BATCH x IN_F
                            const _Float16* __restrict__ B,   // NCOLS x IN_F (col-major B)
                            float* __restrict__ C) {          // BATCH x NCOLS
    const int wave = blockIdx.x * (blockDim.x >> 5) + (threadIdx.x >> 5);
    const int lane = threadIdx.x & 31;

    const int MT = BATCH / 16;          // 128 tiles in M
    const int tm = wave % MT;
    const int tn = wave / MT;           // 0..95
    const int m0 = tm * 16;
    const int n0 = tn * 16;

    const int idx = lane & 15;
    const int hi  = lane >> 4;

    v8f acc = {};

    #pragma unroll
    for (int kc = 0; kc < 2; ++kc) {
        const int bk = kc * 32;

        // A fragment: row m0+idx, K = bk + hi*8 + [0..7] and bk+16+hi*8+[0..7]
        const _Float16* ap = A + (size_t)(m0 + idx) * IN_F + bk + hi * 8;
        v8h alo = *(const v8h*)(ap);
        v8h ahi = *(const v8h*)(ap + 16);
        v16h a;
        #pragma unroll
        for (int i = 0; i < 8; ++i) { a[i] = alo[i]; a[i + 8] = ahi[i]; }

        // B fragment: column n0+idx (= gates row), K = bk + hi*8 + [0..7] and bk+16+hi*8+[0..7]
        const _Float16* bp = B + (size_t)(n0 + idx) * IN_F + bk + hi * 8;
        v8h blo = *(const v8h*)(bp);
        v8h bhi = *(const v8h*)(bp + 16);
        v16h b;
        #pragma unroll
        for (int i = 0; i < 8; ++i) { b[i] = blo[i]; b[i + 8] = bhi[i]; }

        acc = __builtin_amdgcn_wmma_f32_16x16x32_f16(
            /*neg_a=*/false, a, /*neg_b=*/false, b,
            /*c_mod=*/(short)0, acc, /*reuse_a=*/false, /*reuse_b=*/false);
    }

    // Store D: lane column n0+idx, rows m0 + hi*8 + r
    const int ncol  = n0 + idx;
    const int mbase = m0 + hi * 8;
    #pragma unroll
    for (int r = 0; r < 8; ++r)
        C[(size_t)(mbase + r) * NCOLS + ncol] = acc[r];
}

// ---------------------------------------------------------------------------
// Kernel 4: per-tree evaluation + output head.
// One block per batch row; thread j handles tree j (j = layer*128 + tree).
// per_tree = sum_k leaf[k] * prod_d (bit_d(k) ? h_d : 1-h_d)
// computed as 6 successive pair-contractions of the leaf vector with
// (1-h_d, h_d) — bit d is contracted at step d (stride-1 pairs first).
// ---------------------------------------------------------------------------
__global__ void tree_kernel(const float* __restrict__ xproj,  // BATCH x NCOLS
                            const float* __restrict__ th,     // TREES_TOTAL*DEPTH
                            const float* __restrict__ leaf,   // TREES_TOTAL*NUM_LEAVES
                            const float* __restrict__ W,      // 1 x TREES_TOTAL
                            const float* __restrict__ bias,   // 1
                            float* __restrict__ out) {        // BATCH
    const int b = blockIdx.x;
    const int j = threadIdx.x;          // 0..255 = (layer, tree)

    const float* xp = xproj + (size_t)b * NCOLS + (size_t)j * DEPTH;
    const float* tp = th + (size_t)j * DEPTH;

    float h[DEPTH];
    #pragma unroll
    for (int d = 0; d < DEPTH; ++d) {
        float t = xp[d] - tp[d];
        h[d] = 1.0f / (1.0f + __expf(-t));
    }

    float v[NUM_LEAVES];
    const float* lp = leaf + (size_t)j * NUM_LEAVES;
    #pragma unroll
    for (int k = 0; k < NUM_LEAVES; ++k) v[k] = lp[k];

    #pragma unroll
    for (int d = 0; d < DEPTH; ++d) {
        const float hd = h[d];
        const float g  = 1.0f - hd;
        const int half = 32 >> d;       // 32,16,8,4,2,1
        #pragma unroll
        for (int i = 0; i < 32; ++i) {
            if (i < half) v[i] = v[2 * i] * g + v[2 * i + 1] * hd;
        }
    }

    float contrib = v[0] * W[j];

    __shared__ float sdata[TREES_TOTAL];
    sdata[j] = contrib;
    __syncthreads();
    #pragma unroll
    for (int st = TREES_TOTAL / 2; st > 0; st >>= 1) {
        if (j < st) sdata[j] += sdata[j + st];
        __syncthreads();
    }
    if (j == 0) out[b] = sdata[0] + bias[0];
}

// ---------------------------------------------------------------------------
// Host-side launcher
// ---------------------------------------------------------------------------
extern "C" void kernel_launch(void* const* d_in, const int* in_sizes, int n_in,
                              void* d_out, int out_size, void* d_ws, size_t ws_size,
                              hipStream_t stream) {
    const float* x     = (const float*)d_in[0];   // (2048, 64)
    const float* fs    = (const float*)d_in[1];   // (2, 128, 6, 64)
    const float* th    = (const float*)d_in[2];   // (2, 128, 6)
    const float* leaf  = (const float*)d_in[3];   // (2, 128, 64)
    const float* gamma = (const float*)d_in[4];   // (64,)
    const float* beta  = (const float*)d_in[5];   // (64,)
    const float* W     = (const float*)d_in[6];   // (1, 256)
    const float* bias  = (const float*)d_in[7];   // (1,)
    float* out = (float*)d_out;                   // (2048, 1)

    char* ws = (char*)d_ws;
    _Float16* xhat   = (_Float16*)ws;                                         // 256 KB
    _Float16* gatesH = (_Float16*)(ws + (size_t)BATCH * IN_F * 2);            // 192 KB
    float*    xproj  = (float*)(ws + (size_t)BATCH * IN_F * 2
                                   + (size_t)NCOLS * IN_F * 2);               // 12 MB

    bn_kernel<<<IN_F, 256, 0, stream>>>(x, gamma, beta, xhat);
    softmax_kernel<<<NCOLS / 8, 256, 0, stream>>>(fs, gatesH);

    // (2048/16) * (1536/16) = 12288 tiles; 8 waves per 256-thread block.
    gemm_kernel<<<(BATCH / 16) * (NCOLS / 16) / 8, 256, 0, stream>>>(xhat, gatesH, xproj);

    tree_kernel<<<BATCH, 256, 0, stream>>>(xproj, th, leaf, W, bias, out);
}
